// ResOrientedMP_54924041781906
// MI455X (gfx1250) — matrix operand order repaired and checked by reading, not compiled, not run on hardware
//
#include <hip/hip_runtime.h>
#include <hip/hip_bf16.h>
#include <stdint.h>

#define N_NODES  20000
#define N_EDGES  640000
#define N_GRAPHS 64
#define HID      128
#define EF       16
#define NLAYERS  4
#define OUTD     20

typedef __attribute__((ext_vector_type(16))) _Float16 v16h;
typedef __attribute__((ext_vector_type(8)))  float    v8f;

union AFrag { v16h h; uint32_t u[8]; };
union BFrag { v16h h; uint4 q[2]; };
union H2U   { _Float16 f[2]; uint32_t u; };

// ---------------- WMMA helpers (wave32, 16x16x32 f16 -> f32) ----------------

__device__ __forceinline__ void zero_acc(v8f* a, int n) {
  for (int i = 0; i < n; ++i)
    for (int r = 0; r < 8; ++r) a[i][r] = 0.f;
}

// A fragment from LDS tile (16 rows x K halfs, row stride lda halfs).
// 16-bit A 16x32 layout: lane g=lane>>4,m=lane&15; vgpr i, half j ->
// k = (i<4?0:16) + (i&3)*2 + g*8 + j   (j consecutive -> one u32 read)
__device__ __forceinline__ void load_afrag(AFrag& a, const _Float16* Alds,
                                           int lda, int kt, int lane) {
  const int g = (lane >> 4) & 1, m = lane & 15;
  const _Float16* base = Alds + m * lda + kt * 32;
#pragma unroll
  for (int i = 0; i < 8; ++i) {
    const int kb = ((i < 4) ? 0 : 16) + (i & 3) * 2 + g * 8;
    a.u[i] = *(const uint32_t*)(base + kb);
  }
}

__device__ __forceinline__ void load_bfrag(BFrag& b, const uint32_t* Bp,
                                           int idx, int lane) {
  const uint4* q = (const uint4*)Bp + ((size_t)idx * 32 + lane) * 2;
  b.q[0] = q[0];
  b.q[1] = q[1];
}

// M=16 tile GEMM; B fragments preloaded in chunks of 8 to pipeline loads.
template <int KT, int NT>
__device__ __forceinline__ void wave_gemm16(const _Float16* Alds, int lda,
                                            const uint32_t* Bp, v8f* acc,
                                            int lane) {
  for (int kt = 0; kt < KT; ++kt) {
    AFrag a;
    load_afrag(a, Alds, lda, kt, lane);
#pragma unroll
    for (int base = 0; base < NT; base += 8) {
      const int c = (NT - base < 8) ? (NT - base) : 8;
      BFrag b[8];
#pragma unroll
      for (int j = 0; j < 8; ++j)
        if (j < c) load_bfrag(b[j], Bp, kt * NT + base + j, lane);
#pragma unroll
      for (int j = 0; j < 8; ++j)
        if (j < c)
          acc[base + j] = __builtin_amdgcn_wmma_f32_16x16x32_f16(
              false, a.h, false, b[j].h, (short)0, acc[base + j], false, false);
    }
  }
}

// M=32 tile GEMM: two A fragments (rows 0-15 and 16-31) share each B
// fragment -> half the B traffic per row; all 8 B frags of a k-chunk are
// loaded up front so WMMAs retire against a descending loadcnt.
template <int KT>
__device__ __forceinline__ void wave_gemm32(const _Float16* A0,
                                            const _Float16* A1, int lda,
                                            const uint32_t* Bp, v8f* acc0,
                                            v8f* acc1, int lane) {
  for (int kt = 0; kt < KT; ++kt) {
    AFrag a0, a1;
    load_afrag(a0, A0, lda, kt, lane);
    load_afrag(a1, A1, lda, kt, lane);
    BFrag b[8];
#pragma unroll
    for (int nt = 0; nt < 8; ++nt) load_bfrag(b[nt], Bp, kt * 8 + nt, lane);
#pragma unroll
    for (int nt = 0; nt < 8; ++nt) {
      acc0[nt] = __builtin_amdgcn_wmma_f32_16x16x32_f16(
          false, a0.h, false, b[nt].h, (short)0, acc0[nt], false, false);
      acc1[nt] = __builtin_amdgcn_wmma_f32_16x16x32_f16(
          false, a1.h, false, b[nt].h, (short)0, acc1[nt], false, false);
    }
  }
}

// acc[nt][r] holds C[m = (lane>>4)*8 + r][n = nt*16 + (lane&15)]
template <int NT>
__device__ __forceinline__ void bias_act(v8f* acc, const float* b, int lane,
                                         bool do_silu) {
  const int n0 = lane & 15;
#pragma unroll
  for (int nt = 0; nt < NT; ++nt) {
    const float bv = b[nt * 16 + n0];
#pragma unroll
    for (int r = 0; r < 8; ++r) {
      float v = acc[nt][r] + bv;
      if (do_silu) v = v / (1.f + __expf(-v));
      acc[nt][r] = v;
    }
  }
}

template <int NT>
__device__ __forceinline__ void acc_to_lds(const v8f* acc, _Float16* O,
                                           int ldo, int lane) {
  const int g = (lane >> 4) & 1, n0 = lane & 15;
#pragma unroll
  for (int nt = 0; nt < NT; ++nt)
#pragma unroll
    for (int r = 0; r < 8; ++r)
      O[(g * 8 + r) * ldo + nt * 16 + n0] = (_Float16)acc[nt][r];
}

// per-row dot(acc_row, wv) + bias, written to out16[row] (16 rows)
template <int NT>
__device__ __forceinline__ void row_dot(const v8f* acc, const float* wv,
                                        float bias, float* out16, int lane) {
  const int n0 = lane & 15;
  float p[8];
#pragma unroll
  for (int r = 0; r < 8; ++r) p[r] = 0.f;
#pragma unroll
  for (int nt = 0; nt < NT; ++nt) {
    const float w = wv[nt * 16 + n0];
#pragma unroll
    for (int r = 0; r < 8; ++r) p[r] += acc[nt][r] * w;
  }
#pragma unroll
  for (int off = 1; off < 16; off <<= 1)
#pragma unroll
    for (int r = 0; r < 8; ++r) p[r] += __shfl_xor(p[r], off, 32);
  if (n0 == 0) {
    const int g = (lane >> 4) & 1;
#pragma unroll
    for (int r = 0; r < 8; ++r) out16[g * 8 + r] = p[r] + bias;
  }
}

// ---------------- utility kernels ----------------

__device__ __forceinline__ uint32_t wang(uint32_t s) {
  s = (s ^ 61u) ^ (s >> 16); s *= 9u; s ^= s >> 4;
  s *= 0x27d4eb2du; s ^= s >> 15; return s;
}

__global__ void noise_kernel(const float* __restrict__ x,
                             const int* __restrict__ batch,
                             float* __restrict__ xc,
                             float* __restrict__ gsum) {
  const int i = blockIdx.x * blockDim.x + threadIdx.x;
  if (i >= N_NODES) return;
  const int b = batch[i];
#pragma unroll
  for (int d = 0; d < 3; ++d) {
    const uint32_t s1 = wang((uint32_t)(i * 3 + d) * 2654435761u + 42u);
    const uint32_t s2 = wang(s1 ^ 0x9E3779B9u);
    const float u1 = (float)(s1 & 0xFFFFFFu) * (1.f / 16777216.f) + 1e-7f;
    const float u2 = (float)(s2 & 0xFFFFFFu) * (1.f / 16777216.f);
    const float g = sqrtf(-2.f * logf(u1)) * cosf(6.28318530718f * u2);
    const float v = x[i * 3 + d] + 0.01f * g;
    xc[i * 3 + d] = v;
    atomicAdd(&gsum[b * 4 + d], v);
  }
  atomicAdd(&gsum[b * 4 + 3], 1.f);
}

__global__ void center_kernel(float* __restrict__ xc,
                              const int* __restrict__ batch,
                              const float* __restrict__ gsum) {
  const int i = blockIdx.x * blockDim.x + threadIdx.x;
  if (i >= N_NODES) return;
  const int b = batch[i];
  const float c = fmaxf(gsum[b * 4 + 3], 1.f);
#pragma unroll
  for (int d = 0; d < 3; ++d) xc[i * 3 + d] -= gsum[b * 4 + d] / c;
}

__global__ void embed_kernel(const int* __restrict__ h,
                             const float* __restrict__ emb,
                             float* __restrict__ hc,
                             _Float16* __restrict__ hcf) {
  const int idx = blockIdx.x * blockDim.x + threadIdx.x;
  if (idx >= N_NODES * HID) return;
  const int i = idx >> 7, c = idx & 127;
  const float v = emb[h[i] * HID + c];
  hc[idx] = v;
  hcf[idx] = (_Float16)v;
}

// pack f32 row-major [K][N] weight into WMMA B fragments (f16)
__global__ void pack_w_kernel(const float* __restrict__ w, int K, int N,
                              uint32_t* __restrict__ dst, int kTiles,
                              int nTiles) {
  const int t = blockIdx.x * blockDim.x + threadIdx.x;
  const int total = kTiles * nTiles * 256;
  if (t >= total) return;
  const int i = t & 7;
  const int lane = (t >> 3) & 31;
  const int rest = t >> 8;
  const int nt = rest % nTiles;
  const int kt = rest / nTiles;
  const int g = lane >> 4, n0 = lane & 15;
  const int n = nt * 16 + n0;
  H2U pk;
#pragma unroll
  for (int j = 0; j < 2; ++j) {
    const int kk = kt * 32 + ((i < 4) ? 0 : 16) + (i & 3) * 2 + g * 8 + j;  // B 32x16 layout matching A-fragment k order
    const float v = (kk < K && n < N) ? w[kk * N + n] : 0.f;
    pk.f[j] = (_Float16)v;
  }
  dst[t] = pk.u;
}

__global__ void gram_kernel(const float* __restrict__ v1,
                            const float* __restrict__ v2,
                            float* __restrict__ ori) {
  const int i = blockIdx.x * blockDim.x + threadIdx.x;
  if (i >= N_NODES) return;
  float a0 = v1[i * 3 + 0], a1 = v1[i * 3 + 1], a2 = v1[i * 3 + 2];
  float n1 = fmaxf(sqrtf(a0 * a0 + a1 * a1 + a2 * a2), 1e-12f);
  const float e10 = a0 / n1, e11 = a1 / n1, e12 = a2 / n1;
  float b0 = v2[i * 3 + 0], b1 = v2[i * 3 + 1], b2 = v2[i * 3 + 2];
  const float d = e10 * b0 + e11 * b1 + e12 * b2;
  const float p0 = b0 - d * e10, p1 = b1 - d * e11, p2 = b2 - d * e12;
  const float n2 = fmaxf(sqrtf(p0 * p0 + p1 * p1 + p2 * p2), 1e-12f);
  const float e20 = p0 / n2, e21 = p1 / n2, e22 = p2 / n2;
  const float e30 = e11 * e22 - e12 * e21;
  const float e31 = e12 * e20 - e10 * e22;
  const float e32 = e10 * e21 - e11 * e20;
  // ori[i][r][k] = e_k[r]
  ori[i * 9 + 0] = e10; ori[i * 9 + 1] = e20; ori[i * 9 + 2] = e30;
  ori[i * 9 + 3] = e11; ori[i * 9 + 4] = e21; ori[i * 9 + 5] = e31;
  ori[i * 9 + 6] = e12; ori[i * 9 + 7] = e22; ori[i * 9 + 8] = e32;
}

__global__ void proj_kernel(const float* __restrict__ xc,
                            const float* __restrict__ ori,
                            const int* __restrict__ ei,
                            float* __restrict__ proj) {
  const int e = blockIdx.x * blockDim.x + threadIdx.x;
  if (e >= N_EDGES) return;
  const int rn = ei[e], cn = ei[N_EDGES + e];
  const float r0 = xc[rn * 3 + 0] - xc[cn * 3 + 0];
  const float r1 = xc[rn * 3 + 1] - xc[cn * 3 + 1];
  const float r2 = xc[rn * 3 + 2] - xc[cn * 3 + 2];
  const float* O = &ori[cn * 9];
  proj[e * 3 + 0] = O[0] * r0 + O[3] * r1 + O[6] * r2;
  proj[e * 3 + 1] = O[1] * r0 + O[4] * r1 + O[7] * r2;
  proj[e * 3 + 2] = O[2] * r0 + O[5] * r1 + O[8] * r2;
}

// ---------------- fused WMMA edge kernels (M = 32 edges / wave) ----------------

// OrientationLearner: ef = silu(W1 silu(W0 [emb_r,emb_c,ea] + b0) + b1)
// w1 = V11 . silu(V10 ef + b10) + b11 (same for w2); scatter rel*w into v1/v2
__global__ __launch_bounds__(64) void orient_kernel(
    const _Float16* __restrict__ hcf16, const float* __restrict__ ea,
    const float* __restrict__ xc, const int* __restrict__ ei,
    const uint32_t* __restrict__ pe0, const float* __restrict__ e0b,
    const uint32_t* __restrict__ pe1, const float* __restrict__ e1b,
    const uint32_t* __restrict__ pv1, const float* __restrict__ v10b,
    const float* __restrict__ v11w, const float* __restrict__ v11b,
    const uint32_t* __restrict__ pv2, const float* __restrict__ v20b,
    const float* __restrict__ v21w, const float* __restrict__ v21b,
    float* __restrict__ v1o, float* __restrict__ v2o) {
  __shared__ _Float16 tile[2][32 * 288];
  __shared__ float scal[2][2][32];
  const int wave = threadIdx.x >> 5, lane = threadIdx.x & 31;
  const int half = lane >> 4, sub = lane & 15;
  const int e0 = (blockIdx.x * 2 + wave) * 32;
  _Float16* A = tile[wave];
  // gather [emb[row] | emb[col] | edge_attr | pad] (K = 288), 32 rows
  for (int r2 = 0; r2 < 32; r2 += 2) {
    const int r = r2 + half;
    const int e = e0 + r;
    const int rn = ei[e], cn = ei[N_EDGES + e];
    *(uint4*)(A + r * 288 + sub * 8) = *(const uint4*)(hcf16 + rn * HID + sub * 8);
    *(uint4*)(A + r * 288 + 128 + sub * 8) = *(const uint4*)(hcf16 + cn * HID + sub * 8);
    A[r * 288 + 256 + sub] = (_Float16)ea[e * EF + sub];
    A[r * 288 + 272 + sub] = (_Float16)0.f;
  }
  v8f acc0[8], acc1[8];
  zero_acc(acc0, 8);
  zero_acc(acc1, 8);
  wave_gemm32<9>(A, A + 16 * 288, 288, pe0, acc0, acc1, lane);
  bias_act<8>(acc0, e0b, lane, true);
  bias_act<8>(acc1, e0b, lane, true);
  acc_to_lds<8>(acc0, A, 128, lane);          // h1 rows 0-15
  acc_to_lds<8>(acc1, A + 2048, 128, lane);   // h1 rows 16-31
  zero_acc(acc0, 8);
  zero_acc(acc1, 8);
  wave_gemm32<4>(A, A + 2048, 128, pe1, acc0, acc1, lane);
  bias_act<8>(acc0, e1b, lane, true);  // ef (act_last=True)
  bias_act<8>(acc1, e1b, lane, true);
  _Float16* EFL = A + 4096;
  acc_to_lds<8>(acc0, EFL, 128, lane);
  acc_to_lds<8>(acc1, EFL + 2048, 128, lane);
  zero_acc(acc0, 8);
  zero_acc(acc1, 8);
  wave_gemm32<4>(EFL, EFL + 2048, 128, pv1, acc0, acc1, lane);
  bias_act<8>(acc0, v10b, lane, true);
  bias_act<8>(acc1, v10b, lane, true);
  row_dot<8>(acc0, v11w, v11b[0], scal[wave][0], lane);
  row_dot<8>(acc1, v11w, v11b[0], scal[wave][0] + 16, lane);
  zero_acc(acc0, 8);
  zero_acc(acc1, 8);
  wave_gemm32<4>(EFL, EFL + 2048, 128, pv2, acc0, acc1, lane);
  bias_act<8>(acc0, v20b, lane, true);
  bias_act<8>(acc1, v20b, lane, true);
  row_dot<8>(acc0, v21w, v21b[0], scal[wave][1], lane);
  row_dot<8>(acc1, v21w, v21b[0], scal[wave][1] + 16, lane);
  {
    const int e = e0 + lane;  // 32 lanes -> 32 edges
    const int rn = ei[e], cn = ei[N_EDGES + e];
    const float w1 = scal[wave][0][lane], w2 = scal[wave][1][lane];
    const float rx = xc[cn * 3 + 0] - xc[rn * 3 + 0];
    const float ry = xc[cn * 3 + 1] - xc[rn * 3 + 1];
    const float rz = xc[cn * 3 + 2] - xc[rn * 3 + 2];
    atomicAdd(&v1o[rn * 3 + 0], rx * w1);
    atomicAdd(&v1o[rn * 3 + 1], ry * w1);
    atomicAdd(&v1o[rn * 3 + 2], rz * w1);
    atomicAdd(&v2o[rn * 3 + 0], rx * w2);
    atomicAdd(&v2o[rn * 3 + 1], ry * w2);
    atomicAdd(&v2o[rn * 3 + 2], rz * w2);
  }
}

// MP message: m = W1 silu(W0 [hc_c,hc_r,ea,proj] + b0) + b1;
// m *= sigmoid(att.m); aggr[col] += m  (atomic scatter)
__global__ __launch_bounds__(64) void msg_kernel(
    const _Float16* __restrict__ hcf16, const float* __restrict__ ea,
    const float* __restrict__ proj, const int* __restrict__ ei,
    const uint32_t* __restrict__ pm0, const float* __restrict__ m0b,
    const uint32_t* __restrict__ pm1, const float* __restrict__ m1b,
    const float* __restrict__ attw, const float* __restrict__ attb,
    float* __restrict__ aggr) {
  __shared__ _Float16 tile[2][32 * 288];
  __shared__ float scal[2][32];
  __shared__ int cns[2][32];
  const int wave = threadIdx.x >> 5, lane = threadIdx.x & 31;
  const int half = lane >> 4, sub = lane & 15;
  const int e0 = (blockIdx.x * 2 + wave) * 32;
  _Float16* A = tile[wave];
  // gather [hc[col] | hc[row] | ea | proj | pad] (K = 288, valid 275), 32 rows
  for (int r2 = 0; r2 < 32; r2 += 2) {
    const int r = r2 + half;
    const int e = e0 + r;
    const int rn = ei[e], cn = ei[N_EDGES + e];
    *(uint4*)(A + r * 288 + sub * 8) = *(const uint4*)(hcf16 + cn * HID + sub * 8);
    *(uint4*)(A + r * 288 + 128 + sub * 8) = *(const uint4*)(hcf16 + rn * HID + sub * 8);
    A[r * 288 + 256 + sub] = (_Float16)ea[e * EF + sub];
    const float pv = (sub < 3) ? proj[e * 3 + sub] : 0.f;
    A[r * 288 + 272 + sub] = (_Float16)pv;
    if (sub == 0) cns[wave][r] = cn;
  }
  v8f acc0[8], acc1[8];
  zero_acc(acc0, 8);
  zero_acc(acc1, 8);
  wave_gemm32<9>(A, A + 16 * 288, 288, pm0, acc0, acc1, lane);
  bias_act<8>(acc0, m0b, lane, true);
  bias_act<8>(acc1, m0b, lane, true);
  acc_to_lds<8>(acc0, A, 128, lane);
  acc_to_lds<8>(acc1, A + 2048, 128, lane);
  zero_acc(acc0, 8);
  zero_acc(acc1, 8);
  wave_gemm32<4>(A, A + 2048, 128, pm1, acc0, acc1, lane);
  bias_act<8>(acc0, m1b, lane, false);  // m (no final act)
  bias_act<8>(acc1, m1b, lane, false);
  row_dot<8>(acc0, attw, attb[0], scal[wave], lane);
  row_dot<8>(acc1, attw, attb[0], scal[wave] + 16, lane);
  const int g = half, n0 = sub;
#pragma unroll
  for (int part = 0; part < 2; ++part) {
    const v8f* acc = part ? acc1 : acc0;
#pragma unroll
    for (int r = 0; r < 8; ++r) {
      const int m = part * 16 + g * 8 + r;
      const float gate = 1.f / (1.f + __expf(-scal[wave][m]));
      const int cn = cns[wave][m];
#pragma unroll
      for (int nt = 0; nt < 8; ++nt)
        atomicAdd(&aggr[cn * HID + nt * 16 + n0], acc[nt][r] * gate);
    }
  }
}

// MP update: hc += U1 silu(U0 [hc,aggr] + b0) + b1
__global__ __launch_bounds__(128) void upd_kernel(
    const float* __restrict__ aggr, const uint32_t* __restrict__ pu0,
    const float* __restrict__ u0b, const uint32_t* __restrict__ pu1,
    const float* __restrict__ u1b, float* __restrict__ hc,
    _Float16* __restrict__ hcf16) {
  __shared__ _Float16 tile[4][16 * 256];
  const int wave = threadIdx.x >> 5, lane = threadIdx.x & 31;
  const int half = lane >> 4, sub = lane & 15;
  const int t = blockIdx.x * 4 + wave;
  if (t >= N_NODES / 16) return;
  const int n0node = t * 16;
  _Float16* A = tile[wave];
  for (int r2 = 0; r2 < 16; r2 += 2) {
    const int r = r2 + half;
    const int node = n0node + r;
    *(uint4*)(A + r * 256 + sub * 8) = *(const uint4*)(hcf16 + node * HID + sub * 8);
#pragma unroll
    for (int q = 0; q < 8; ++q)
      A[r * 256 + 128 + sub * 8 + q] = (_Float16)aggr[node * HID + sub * 8 + q];
  }
  v8f acc[8];
  zero_acc(acc, 8);
  wave_gemm16<8, 8>(A, 256, pu0, acc, lane);
  bias_act<8>(acc, u0b, lane, true);
  acc_to_lds<8>(acc, A, 128, lane);
  zero_acc(acc, 8);
  wave_gemm16<4, 8>(A, 128, pu1, acc, lane);
  const int g = half, c0 = sub;
#pragma unroll
  for (int nt = 0; nt < 8; ++nt)
#pragma unroll
    for (int r = 0; r < 8; ++r) {
      const int node = n0node + g * 8 + r;
      const int col = nt * 16 + c0;
      const float v = hc[node * HID + col] + acc[nt][r] + u1b[col];
      hc[node * HID + col] = v;
      hcf16[node * HID + col] = (_Float16)v;
    }
}

// out = O1 silu(O0 hc + b0) + b1   (128 -> 256 -> 20)
__global__ __launch_bounds__(128) void out_kernel(
    const _Float16* __restrict__ hcf16, const uint32_t* __restrict__ po0,
    const float* __restrict__ o0b, const uint32_t* __restrict__ po1,
    const float* __restrict__ o1b, float* __restrict__ out) {
  __shared__ _Float16 tile[4][16 * 256];
  const int wave = threadIdx.x >> 5, lane = threadIdx.x & 31;
  const int half = lane >> 4, sub = lane & 15;
  const int t = blockIdx.x * 4 + wave;
  if (t >= N_NODES / 16) return;
  const int n0node = t * 16;
  _Float16* A = tile[wave];
  for (int r2 = 0; r2 < 16; r2 += 2) {
    const int r = r2 + half;
    const int node = n0node + r;
    *(uint4*)(A + r * 128 + sub * 8) = *(const uint4*)(hcf16 + node * HID + sub * 8);
  }
  v8f acc[16];
  zero_acc(acc, 16);
  wave_gemm16<4, 16>(A, 128, po0, acc, lane);
  bias_act<16>(acc, o0b, lane, true);
  acc_to_lds<16>(acc, A, 256, lane);  // 16x256 intermediate (overwrites input)
  v8f acc2[2];
  zero_acc(acc2, 2);
  wave_gemm16<8, 2>(A, 256, po1, acc2, lane);
  const int g = half, c0 = sub;
#pragma unroll
  for (int nt = 0; nt < 2; ++nt)
#pragma unroll
    for (int r = 0; r < 8; ++r) {
      const int col = nt * 16 + c0;
      if (col < OUTD) {
        const int node = n0node + g * 8 + r;
        out[node * OUTD + col] = acc2[nt][r] + o1b[col];
      }
    }
}

// ---------------- host launcher ----------------

extern "C" void kernel_launch(void* const* d_in, const int* in_sizes, int n_in,
                              void* d_out, int out_size, void* d_ws,
                              size_t ws_size, hipStream_t stream) {
  (void)in_sizes; (void)n_in; (void)out_size; (void)ws_size;
  // inputs (jax tree-flatten order: dict keys sorted)
  const int* h = (const int*)d_in[0];
  const float* x = (const float*)d_in[1];
  const int* ei = (const int*)d_in[2];
  const float* ea = (const float*)d_in[3];
  const int* batch = (const int*)d_in[4];
  const float* embedW = (const float*)d_in[5];
  struct Layer {
    const float *attb, *attw, *m0b, *m0w, *m1b, *m1w, *u0b, *u0w, *u1b, *u1w;
  } L[NLAYERS];
  int k = 6;
  for (int l = 0; l < NLAYERS; ++l) {
    L[l].attb = (const float*)d_in[k + 0];
    L[l].attw = (const float*)d_in[k + 1];
    L[l].m0b = (const float*)d_in[k + 2];
    L[l].m0w = (const float*)d_in[k + 3];
    L[l].m1b = (const float*)d_in[k + 4];
    L[l].m1w = (const float*)d_in[k + 5];
    L[l].u0b = (const float*)d_in[k + 6];
    L[l].u0w = (const float*)d_in[k + 7];
    L[l].u1b = (const float*)d_in[k + 8];
    L[l].u1w = (const float*)d_in[k + 9];
    k += 10;
  }
  const float* e0b = (const float*)d_in[46];
  const float* e0w = (const float*)d_in[47];
  const float* e1b = (const float*)d_in[48];
  const float* e1w = (const float*)d_in[49];
  const float* v10b = (const float*)d_in[50];
  const float* v10w = (const float*)d_in[51];
  const float* v11b = (const float*)d_in[52];
  const float* v11w = (const float*)d_in[53];
  const float* v20b = (const float*)d_in[54];
  const float* v20w = (const float*)d_in[55];
  const float* v21b = (const float*)d_in[56];
  const float* v21w = (const float*)d_in[57];
  const float* o0b = (const float*)d_in[58];
  const float* o0w = (const float*)d_in[59];
  const float* o1b = (const float*)d_in[60];
  const float* o1w = (const float*)d_in[61];
  float* out = (float*)d_out;

  // workspace carve-out
  char* w = (char*)d_ws;
  auto alloc = [&](size_t bytes) -> void* {
    void* p = (void*)w;
    w += (bytes + 255) & ~(size_t)255;
    return p;
  };
  float* xc = (float*)alloc((size_t)N_NODES * 3 * 4);
  float* gsum = (float*)alloc((size_t)N_GRAPHS * 4 * 4);
  float* hc = (float*)alloc((size_t)N_NODES * HID * 4);
  _Float16* hcf = (_Float16*)alloc((size_t)N_NODES * HID * 2);
  float* v1 = (float*)alloc((size_t)N_NODES * 3 * 4);
  float* v2 = (float*)alloc((size_t)N_NODES * 3 * 4);
  float* ori = (float*)alloc((size_t)N_NODES * 9 * 4);
  float* proj = (float*)alloc((size_t)N_EDGES * 3 * 4);
  float* aggr = (float*)alloc((size_t)N_NODES * HID * 4);
  // packed weights (uint32 counts = kTiles*nTiles*256)
  uint32_t* pe0 = (uint32_t*)alloc((size_t)9 * 8 * 256 * 4);
  uint32_t* pe1 = (uint32_t*)alloc((size_t)4 * 8 * 256 * 4);
  uint32_t* pv1 = (uint32_t*)alloc((size_t)4 * 8 * 256 * 4);
  uint32_t* pv2 = (uint32_t*)alloc((size_t)4 * 8 * 256 * 4);
  uint32_t* pm0[NLAYERS];
  uint32_t* pm1[NLAYERS];
  uint32_t* pu0[NLAYERS];
  uint32_t* pu1[NLAYERS];
  for (int l = 0; l < NLAYERS; ++l) {
    pm0[l] = (uint32_t*)alloc((size_t)9 * 8 * 256 * 4);
    pm1[l] = (uint32_t*)alloc((size_t)4 * 8 * 256 * 4);
    pu0[l] = (uint32_t*)alloc((size_t)8 * 8 * 256 * 4);
    pu1[l] = (uint32_t*)alloc((size_t)4 * 8 * 256 * 4);
  }
  uint32_t* po0 = (uint32_t*)alloc((size_t)4 * 16 * 256 * 4);
  uint32_t* po1 = (uint32_t*)alloc((size_t)8 * 2 * 256 * 4);

  auto pack = [&](const float* wsrc, int K, int N, int padN, uint32_t* dst) {
    const int kT = (K + 31) / 32, nT = padN / 16;
    const int total = kT * nT * 256;
    pack_w_kernel<<<(total + 255) / 256, 256, 0, stream>>>(wsrc, K, N, dst, kT, nT);
  };

  // 1) noise + per-graph centering
  hipMemsetAsync(gsum, 0, (size_t)N_GRAPHS * 4 * 4, stream);
  noise_kernel<<<(N_NODES + 255) / 256, 256, 0, stream>>>(x, batch, xc, gsum);
  center_kernel<<<(N_NODES + 255) / 256, 256, 0, stream>>>(xc, batch, gsum);

  // 2) node embeddings (hc = emb[h]) + f16 copy
  embed_kernel<<<(N_NODES * HID + 255) / 256, 256, 0, stream>>>(h, embedW, hc, hcf);

  // 3) pack all GEMM weights into WMMA B-fragment order
  pack(e0w, 272, HID, HID, pe0);
  pack(e1w, HID, HID, HID, pe1);
  pack(v10w, HID, HID, HID, pv1);
  pack(v20w, HID, HID, HID, pv2);
  for (int l = 0; l < NLAYERS; ++l) {
    pack(L[l].m0w, 275, HID, HID, pm0[l]);
    pack(L[l].m1w, HID, HID, HID, pm1[l]);
    pack(L[l].u0w, 256, HID, HID, pu0[l]);
    pack(L[l].u1w, HID, HID, HID, pu1[l]);
  }
  pack(o0w, HID, 256, 256, po0);
  pack(o1w, 256, OUTD, 32, po1);

  // 4) orientation learner (32 edges/wave, 2 waves/block)
  hipMemsetAsync(v1, 0, (size_t)N_NODES * 3 * 4, stream);
  hipMemsetAsync(v2, 0, (size_t)N_NODES * 3 * 4, stream);
  orient_kernel<<<N_EDGES / 64, 64, 0, stream>>>(
      hcf, ea, xc, ei, pe0, e0b, pe1, e1b, pv1, v10b, v11w, v11b, pv2, v20b,
      v21w, v21b, v1, v2);
  gram_kernel<<<(N_NODES + 255) / 256, 256, 0, stream>>>(v1, v2, ori);
  proj_kernel<<<(N_EDGES + 255) / 256, 256, 0, stream>>>(xc, ori, ei, proj);

  // 5) message-passing layers
  const int nodeBlocks = (N_NODES / 16 + 3) / 4;
  for (int l = 0; l < NLAYERS; ++l) {
    hipMemsetAsync(aggr, 0, (size_t)N_NODES * HID * 4, stream);
    msg_kernel<<<N_EDGES / 64, 64, 0, stream>>>(
        hcf, ea, proj, ei, pm0[l], L[l].m0b, pm1[l], L[l].m1b, L[l].attw,
        L[l].attb, aggr);
    upd_kernel<<<nodeBlocks, 128, 0, stream>>>(aggr, pu0[l], L[l].u0b, pu1[l],
                                               L[l].u1b, hc, hcf);
  }

  // 6) output head
  out_kernel<<<nodeBlocks, 128, 0, stream>>>(hcf, po0, o0b, po1, o1b, out);
}
